// TransformerAggregator_32959579030270
// MI455X (gfx1250) — compile-verified, matrix operand order checked by reading, not run on hardware
//
#include <hip/hip_runtime.h>
#include <cstdint>

// CDNA5 / gfx1250: bf16 WMMA GEMMs (double-buffered, reg-staged, prefetching)
// + fused rel-shift causal flash attention.
typedef __bf16 bf16_t;
typedef __bf16 v16bf __attribute__((ext_vector_type(16)));
typedef float  v8f   __attribute__((ext_vector_type(8)));

#define QLEN 1024
#define BSZ  4
#define DMODEL 512
#define NH   8
#define DH   64
#define FFD  2048
#define PLEN 2048
#define SEQ  (QLEN * BSZ)

__device__ __forceinline__ v8f wmma_bf16(v16bf a, v16bf b, v8f c) {
  return __builtin_amdgcn_wmma_f32_16x16x32_bf16(false, a, false, b, (short)0, c,
                                                 false, false);
}

// Load a 16x32 bf16 tile (row-major, `pitch` elements between rows) into the
// CDNA5 WMMA A-operand layout (also used for B via B^T).
// lanes 0-15: M=lane, K in {0..7,16..23}; lanes 16-31: M=lane-16, K+8.
__device__ __forceinline__ v16bf load_frag(const bf16_t* p, int pitch) {
  const int lane = threadIdx.x & 31;
  const bf16_t* row = p + (size_t)(lane & 15) * pitch + (lane >> 4) * 8;
  v16bf f;
#pragma unroll
  for (int g = 0; g < 8; ++g) {
    const int k = (g < 4) ? (2 * g) : (2 * g + 8);
    f[2 * g]     = row[k];
    f[2 * g + 1] = row[k + 1];
  }
  return f;
}

__device__ __forceinline__ float wave_sum(float v) {
#pragma unroll
  for (int m = 16; m >= 1; m >>= 1) v += __shfl_xor(v, m);
  return v;
}
__device__ __forceinline__ float half_max(float v) {  // across a 16-lane half
#pragma unroll
  for (int m = 8; m >= 1; m >>= 1) v = fmaxf(v, __shfl_xor(v, m));
  return v;
}
__device__ __forceinline__ float half_sum(float v) {
#pragma unroll
  for (int m = 8; m >= 1; m >>= 1) v += __shfl_xor(v, m);
  return v;
}

// ---------------------------------------------------------------------------
// Positional encoding: pe[p][c] = sin((1024-p)*invfreq[c]) | cos (c>=256)
// ---------------------------------------------------------------------------
__global__ __launch_bounds__(256) void posemb_kernel(float* __restrict__ pe) {
  const int p = blockIdx.x;
  const int c = threadIdx.x;  // 0..255 -> sin col c, cos col c+256
  const float pos = (float)(QLEN - p);
  const float inv = __expf(-(float)c * (9.210340371976184f / 256.0f));  // ln(1e4)/256
  const float a = pos * inv;
  pe[(size_t)p * DMODEL + c]       = sinf(a);
  pe[(size_t)p * DMODEL + c + 256] = cosf(a);
}

// ---------------------------------------------------------------------------
// bf16 WMMA GEMM: C = A @ B (or A @ B^T). Block tile 128x128, 8 waves (4x2),
// wave tile 32x64 (2x4 of 16x16), K-step 32. Double-buffered LDS with
// register-staged global loads (next tile fetched under current WMMAs) and
// global_prefetch two tiles ahead.
// ---------------------------------------------------------------------------
enum { M_F32 = 0, M_F32_BIAS, M_GELU_BF16, M_QDUAL, M_KHEAD, M_VHEADT, M_KPOS };

#define GBM 128
#define GBN 128
#define GBK 32
#define APITCH 40  // halfwords per LDS row (16B-aligned rows, odd 16B-unit pitch)

template <int MODE, bool ABF16, bool BT>
__global__ __launch_bounds__(256) void gemm_kernel(
    const void* __restrict__ Ap, const float* __restrict__ Bp,
    const float* __restrict__ bias0, const float* __restrict__ bias1,
    void* __restrict__ out0, void* __restrict__ out1, const int N, const int K) {
  __shared__ __align__(16) bf16_t Asm[2][GBM * APITCH];
  __shared__ __align__(16) bf16_t Bsm[2][GBN * APITCH];

  const int tid = threadIdx.x;
  const int wave = tid >> 5;
  const int wr = wave >> 1, wc = wave & 1;
  const int by = blockIdx.y * GBM;
  const int bx = blockIdx.x * GBN;

  uint4  aRb[2];   // bf16 A staging (raw copy)
  float4 aRf[4];   // f32 A staging
  float4 bRf[4];   // B^T staging (f32)
  float  bRs[16];  // B (K-major) transpose staging

  auto load_tile = [&](int k0) {
    if constexpr (ABF16) {
      const bf16_t* A = (const bf16_t*)Ap;
#pragma unroll
      for (int i = 0; i < 2; ++i) {
        const int idx = tid + i * 256;
        aRb[i] = *reinterpret_cast<const uint4*>(
            A + (size_t)(by + (idx >> 2)) * K + k0 + (idx & 3) * 8);
      }
    } else {
      const float* A = (const float*)Ap;
#pragma unroll
      for (int i = 0; i < 4; ++i) {
        const int idx = tid + i * 256;
        aRf[i] = *reinterpret_cast<const float4*>(
            A + (size_t)(by + (idx >> 3)) * K + k0 + (idx & 7) * 4);
      }
    }
    if constexpr (BT) {  // B is [N x K] row-major
#pragma unroll
      for (int i = 0; i < 4; ++i) {
        const int idx = tid + i * 256;
        bRf[i] = *reinterpret_cast<const float4*>(
            Bp + (size_t)(bx + (idx >> 3)) * K + k0 + (idx & 7) * 4);
      }
    } else {  // B is [K x N] row-major: gather for transpose
#pragma unroll
      for (int i = 0; i < 16; ++i) {
        const int flat = tid + i * 256;  // 32 k x 128 n
        bRs[i] = Bp[(size_t)(k0 + (flat >> 7)) * N + bx + (flat & 127)];
      }
    }
  };
  auto store_tile = [&](int buf) {
    if constexpr (ABF16) {
#pragma unroll
      for (int i = 0; i < 2; ++i) {
        const int idx = tid + i * 256;
        *reinterpret_cast<uint4*>(&Asm[buf][(idx >> 2) * APITCH + (idx & 3) * 8]) =
            aRb[i];
      }
    } else {
#pragma unroll
      for (int i = 0; i < 4; ++i) {
        const int idx = tid + i * 256;
        bf16_t* d = &Asm[buf][(idx >> 3) * APITCH + (idx & 7) * 4];
        d[0] = (bf16_t)aRf[i].x; d[1] = (bf16_t)aRf[i].y;
        d[2] = (bf16_t)aRf[i].z; d[3] = (bf16_t)aRf[i].w;
      }
    }
    if constexpr (BT) {
#pragma unroll
      for (int i = 0; i < 4; ++i) {
        const int idx = tid + i * 256;
        bf16_t* d = &Bsm[buf][(idx >> 3) * APITCH + (idx & 7) * 4];
        d[0] = (bf16_t)bRf[i].x; d[1] = (bf16_t)bRf[i].y;
        d[2] = (bf16_t)bRf[i].z; d[3] = (bf16_t)bRf[i].w;
      }
    } else {
#pragma unroll
      for (int i = 0; i < 16; ++i) {
        const int flat = tid + i * 256;
        Bsm[buf][(flat & 127) * APITCH + (flat >> 7)] = (bf16_t)bRs[i];
      }
    }
  };

  v8f acc[2][4] = {};

  const int KT = K / GBK;
  load_tile(0);
  store_tile(0);
  __syncthreads();
  for (int t = 0; t < KT; ++t) {
    if (t + 1 < KT) {
      load_tile((t + 1) * GBK);  // overlaps with WMMAs below
      if (t + 2 < KT) {          // warm GL2 two tiles ahead
        const int k2 = (t + 2) * GBK;
        if constexpr (ABF16) {
          const bf16_t* A = (const bf16_t*)Ap;
          __builtin_prefetch(A + (size_t)(by + (tid >> 1)) * K + k2 + (tid & 1) * 16,
                             0, 1);
        } else {
          const float* A = (const float*)Ap;
          __builtin_prefetch(A + (size_t)(by + (tid >> 1)) * K + k2 + (tid & 1) * 16,
                             0, 1);
        }
      }
    }
    const int buf = t & 1;
    const bf16_t* aB = &Asm[buf][(wr * 32) * APITCH];
    const bf16_t* bB = &Bsm[buf][(wc * 64) * APITCH];
    const v16bf a0 = load_frag(aB, APITCH);
    const v16bf a1 = load_frag(aB + 16 * APITCH, APITCH);
#pragma unroll
    for (int ni = 0; ni < 4; ++ni) {
      const v16bf bfr = load_frag(bB + ni * 16 * APITCH, APITCH);
      acc[0][ni] = wmma_bf16(a0, bfr, acc[0][ni]);
      acc[1][ni] = wmma_bf16(a1, bfr, acc[1][ni]);
    }
    if (t + 1 < KT) store_tile((t + 1) & 1);  // other buffer: no hazard
    __syncthreads();
  }

  const int lane = tid & 31;
  const int cc = lane & 15, hh = lane >> 4;
#pragma unroll
  for (int mi = 0; mi < 2; ++mi)
#pragma unroll
    for (int ni = 0; ni < 4; ++ni)
#pragma unroll
      for (int r = 0; r < 8; ++r) {
        const int row = by + wr * 32 + mi * 16 + r + 8 * hh;
        const int col = bx + wc * 64 + ni * 16 + cc;
        const float v = acc[mi][ni][r];
        if constexpr (MODE == M_F32) {
          ((float*)out0)[(size_t)row * N + col] = v;
        } else if constexpr (MODE == M_F32_BIAS) {
          ((float*)out0)[(size_t)row * N + col] = v + bias0[col];
        } else if constexpr (MODE == M_GELU_BF16) {
          float t = v + bias0[col];
          t = 0.5f * t * (1.0f + erff(t * 0.70710678118654752f));  // exact GELU
          ((bf16_t*)out0)[(size_t)row * N + col] = (bf16_t)t;
        } else if constexpr (MODE == M_QDUAL) {
          const int i = row >> 2, bb = row & 3, nn = col >> 6, dd = col & 63;
          const size_t o = (((size_t)bb * NH + nn) * QLEN + i) * DH + dd;
          ((bf16_t*)out0)[o] = (bf16_t)(v + bias0[col]);  // Q + r_w_bias
          ((bf16_t*)out1)[o] = (bf16_t)(v + bias1[col]);  // Q + r_r_bias
        } else if constexpr (MODE == M_KHEAD) {
          const int i = row >> 2, bb = row & 3, nn = col >> 6, dd = col & 63;
          ((bf16_t*)out0)[(((size_t)bb * NH + nn) * QLEN + i) * DH + dd] = (bf16_t)v;
        } else if constexpr (MODE == M_VHEADT) {
          const int i = row >> 2, bb = row & 3, nn = col >> 6, dd = col & 63;
          ((bf16_t*)out0)[(((size_t)bb * NH + nn) * DH + dd) * QLEN + i] = (bf16_t)v;
        } else if constexpr (MODE == M_KPOS) {
          const int nn = col >> 6, dd = col & 63;
          ((bf16_t*)out0)[((size_t)nn * PLEN + row) * DH + dd] = (bf16_t)v;
        }
      }
}

// ---------------------------------------------------------------------------
// Fused causal flash attention with XLNet rel-shift.
// bd_shifted[i,j] = Qbd[i] . Kpos[1024 + j - i]  -> computed as a 16x48 band.
// Grid: (QLEN/64, NH, BSZ); 4 waves, each wave owns 16 query rows.
// ---------------------------------------------------------------------------
__global__ __launch_bounds__(128) void attn_kernel(
    const bf16_t* __restrict__ Qac, const bf16_t* __restrict__ Qbd,
    const bf16_t* __restrict__ Kb, const bf16_t* __restrict__ Vt,
    const bf16_t* __restrict__ Kpos, bf16_t* __restrict__ av) {
  const int n = blockIdx.y, b = blockIdx.z;
  const int wave = threadIdx.x >> 5, lane = threadIdx.x & 31;
  const int cc = lane & 15, hh = lane >> 4;
  const int i0 = blockIdx.x * 64 + wave * 16;

  const size_t headQ = ((size_t)b * NH + n) * QLEN * DH;
  const bf16_t* Qacp = Qac + headQ;
  const bf16_t* Qbdp = Qbd + headQ;
  const bf16_t* Kbp  = Kb + headQ;
  const bf16_t* Vtp  = Vt + headQ;                    // [DH][QLEN]
  const bf16_t* Kpp  = Kpos + (size_t)n * PLEN * DH;  // [PLEN][DH]

  const v16bf qa0 = load_frag(Qacp + (size_t)i0 * DH, DH);
  const v16bf qa1 = load_frag(Qacp + (size_t)i0 * DH + 32, DH);
  const v16bf qb0 = load_frag(Qbdp + (size_t)i0 * DH, DH);
  const v16bf qb1 = load_frag(Qbdp + (size_t)i0 * DH + 32, DH);

  __shared__ float  Tld[4][16][49];   // wave-private bd band
  __shared__ bf16_t Pld[4][16][34];   // wave-private P tile (A-operand staging)

  float m_r[8], l_r[8];
#pragma unroll
  for (int r = 0; r < 8; ++r) { m_r[r] = -3.0e38f; l_r[r] = 0.0f; }
  v8f o0 = {}, o1 = {}, o2 = {}, o3 = {};

  const float scale = 0.125f;  // 1/sqrt(64)
  const int jbmax = (i0 + 15) >> 5;
  for (int jb = 0; jb <= jbmax; ++jb) {
    const int j0 = jb * 32;
    if (jb < jbmax)  // warm GL2 with next K block (32 rows x 64 = lane*128B)
      __builtin_prefetch(Kbp + (size_t)(j0 + 32) * DH + lane * 64, 0, 1);
    // ---- ac scores: Qac[16x64] @ K[32x64]^T -> 2 tiles of 16x16
    v8f sac0 = {}, sac1 = {};
    {
      const v16bf k00 = load_frag(Kbp + (size_t)j0 * DH, DH);
      const v16bf k01 = load_frag(Kbp + (size_t)j0 * DH + 32, DH);
      const v16bf k10 = load_frag(Kbp + (size_t)(j0 + 16) * DH, DH);
      const v16bf k11 = load_frag(Kbp + (size_t)(j0 + 16) * DH + 32, DH);
      sac0 = wmma_bf16(qa0, k00, sac0);
      sac0 = wmma_bf16(qa1, k01, sac0);
      sac1 = wmma_bf16(qa0, k10, sac1);
      sac1 = wmma_bf16(qa1, k11, sac1);
    }
    // ---- bd band: T[a,u] = Qbd[i0+a] . Kpos[pbase+u], u in [0,48)
    const int pbase = QLEN + j0 - i0 - 15;  // always within [1, 2048-48]
    v8f tt[3] = {};
#pragma unroll
    for (int ut = 0; ut < 3; ++ut) {
      const v16bf p0 = load_frag(Kpp + (size_t)(pbase + ut * 16) * DH, DH);
      const v16bf p1 = load_frag(Kpp + (size_t)(pbase + ut * 16) * DH + 32, DH);
      tt[ut] = wmma_bf16(qb0, p0, tt[ut]);
      tt[ut] = wmma_bf16(qb1, p1, tt[ut]);
    }
#pragma unroll
    for (int ut = 0; ut < 3; ++ut)
#pragma unroll
      for (int r = 0; r < 8; ++r)
        Tld[wave][r + 8 * hh][ut * 16 + cc] = tt[ut][r];

    // ---- gather shift, mask, streaming softmax
#pragma unroll
    for (int r = 0; r < 8; ++r) {
      const int a = r + 8 * hh;
      const int ig = i0 + a;
      float s0 = (sac0[r] + Tld[wave][a][cc + 15 - a]) * scale;
      float s1 = (sac1[r] + Tld[wave][a][cc + 31 - a]) * scale;
      if (j0 + cc > ig)      s0 = -3.0e38f;  // causal mask
      if (j0 + 16 + cc > ig) s1 = -3.0e38f;
      const float mx = half_max(fmaxf(s0, s1));
      const float mnew = fmaxf(m_r[r], mx);
      const float e0 = __expf(s0 - mnew);
      const float e1 = __expf(s1 - mnew);
      const float rs = half_sum(e0 + e1);
      const float alpha = __expf(m_r[r] - mnew);
      l_r[r] = l_r[r] * alpha + rs;
      m_r[r] = mnew;
      o0[r] = o0[r] * alpha; o1[r] = o1[r] * alpha;
      o2[r] = o2[r] * alpha; o3[r] = o3[r] * alpha;
      Pld[wave][a][cc]      = (bf16_t)e0;
      Pld[wave][a][16 + cc] = (bf16_t)e1;
    }
    // ---- O += P[16x32] @ V[32x64]
    const v16bf pf = load_frag(&Pld[wave][0][0], 34);
    const v16bf v0 = load_frag(Vtp + (size_t)0  * QLEN + j0, QLEN);
    const v16bf v1 = load_frag(Vtp + (size_t)16 * QLEN + j0, QLEN);
    const v16bf v2 = load_frag(Vtp + (size_t)32 * QLEN + j0, QLEN);
    const v16bf v3 = load_frag(Vtp + (size_t)48 * QLEN + j0, QLEN);
    o0 = wmma_bf16(pf, v0, o0);
    o1 = wmma_bf16(pf, v1, o1);
    o2 = wmma_bf16(pf, v2, o2);
    o3 = wmma_bf16(pf, v3, o3);
  }

#pragma unroll
  for (int r = 0; r < 8; ++r) {
    const float inv = 1.0f / l_r[r];
    const int ig = i0 + r + 8 * hh;
    const size_t base = ((size_t)ig * BSZ + b) * DMODEL + n * DH;
    av[base + 0  + cc] = (bf16_t)(o0[r] * inv);
    av[base + 16 + cc] = (bf16_t)(o1[r] * inv);
    av[base + 32 + cc] = (bf16_t)(o2[r] * inv);
    av[base + 48 + cc] = (bf16_t)(o3[r] * inv);
  }
}

// ---------------------------------------------------------------------------
// LayerNorm kernels (512-wide rows; 128 threads, 4 elems/thread)
// ---------------------------------------------------------------------------
__global__ __launch_bounds__(128) void ln_res_kernel(
    const float* __restrict__ a, const float* __restrict__ res,
    const float* __restrict__ g, const float* __restrict__ be,
    float* __restrict__ xo, bf16_t* __restrict__ xb) {
  const int row = blockIdx.x;
  const int t = threadIdx.x;
  __shared__ float r1[4], r2[4];
  float v[4], s = 0.f, s2 = 0.f;
#pragma unroll
  for (int k = 0; k < 4; ++k) {
    const int c = t + k * 128;
    const float x = a[(size_t)row * DMODEL + c] + res[(size_t)row * DMODEL + c];
    v[k] = x; s += x; s2 += x * x;
  }
  s = wave_sum(s); s2 = wave_sum(s2);
  if ((t & 31) == 0) { r1[t >> 5] = s; r2[t >> 5] = s2; }
  __syncthreads();
  s = r1[0] + r1[1] + r1[2] + r1[3];
  s2 = r2[0] + r2[1] + r2[2] + r2[3];
  const float mu = s * (1.0f / DMODEL);
  const float rstd = rsqrtf(s2 * (1.0f / DMODEL) - mu * mu + 1e-5f);
#pragma unroll
  for (int k = 0; k < 4; ++k) {
    const int c = t + k * 128;
    const float y = (v[k] - mu) * rstd * g[c] + be[c];
    xo[(size_t)row * DMODEL + c] = y;
    xb[(size_t)row * DMODEL + c] = (bf16_t)y;
  }
}

__global__ __launch_bounds__(128) void ln2_final_kernel(
    const float* __restrict__ x, const float* __restrict__ y,
    const float* __restrict__ g2, const float* __restrict__ be2,
    const float* __restrict__ gf, const float* __restrict__ bfp,
    float* __restrict__ out) {
  const int row = blockIdx.x;
  const int t = threadIdx.x;
  __shared__ float r1[4], r2[4];
  float v[4], s = 0.f, s2 = 0.f;
#pragma unroll
  for (int k = 0; k < 4; ++k) {
    const int c = t + k * 128;
    const float xv = x[(size_t)row * DMODEL + c] + y[(size_t)row * DMODEL + c];
    v[k] = xv; s += xv; s2 += xv * xv;
  }
  s = wave_sum(s); s2 = wave_sum(s2);
  if ((t & 31) == 0) { r1[t >> 5] = s; r2[t >> 5] = s2; }
  __syncthreads();
  s = r1[0] + r1[1] + r1[2] + r1[3];
  s2 = r2[0] + r2[1] + r2[2] + r2[3];
  float mu = s * (1.0f / DMODEL);
  float rstd = rsqrtf(s2 * (1.0f / DMODEL) - mu * mu + 1e-5f);
  float s3 = 0.f, s4 = 0.f;
#pragma unroll
  for (int k = 0; k < 4; ++k) {
    const int c = t + k * 128;
    const float tv = (v[k] - mu) * rstd * g2[c] + be2[c];
    v[k] = tv; s3 += tv; s4 += tv * tv;
  }
  __syncthreads();  // r1/r2 reuse
  s3 = wave_sum(s3); s4 = wave_sum(s4);
  if ((t & 31) == 0) { r1[t >> 5] = s3; r2[t >> 5] = s4; }
  __syncthreads();
  s3 = r1[0] + r1[1] + r1[2] + r1[3];
  s4 = r2[0] + r2[1] + r2[2] + r2[3];
  mu = s3 * (1.0f / DMODEL);
  rstd = rsqrtf(s4 * (1.0f / DMODEL) - mu * mu + 1e-5f);
#pragma unroll
  for (int k = 0; k < 4; ++k) {
    const int c = t + k * 128;
    out[(size_t)row * DMODEL + c] = (v[k] - mu) * rstd * gf[c] + bfp[c];
  }
}

// ---------------------------------------------------------------------------
extern "C" void kernel_launch(void* const* d_in, const int* in_sizes, int n_in,
                              void* d_out, int out_size, void* d_ws, size_t ws_size,
                              hipStream_t stream) {
  (void)in_sizes; (void)n_in; (void)out_size; (void)ws_size;
  const float* src  = (const float*)d_in[0];
  const float* q_w  = (const float*)d_in[1];
  const float* k_w  = (const float*)d_in[2];
  const float* v_w  = (const float*)d_in[3];
  const float* r_w  = (const float*)d_in[4];
  const float* o_w  = (const float*)d_in[5];
  const float* rwb  = (const float*)d_in[6];
  const float* rrb  = (const float*)d_in[7];
  const float* w1   = (const float*)d_in[8];
  const float* b1   = (const float*)d_in[9];
  const float* w2   = (const float*)d_in[10];
  const float* b2   = (const float*)d_in[11];
  const float* g1   = (const float*)d_in[12];
  const float* be1  = (const float*)d_in[13];
  const float* g2   = (const float*)d_in[14];
  const float* be2  = (const float*)d_in[15];
  const float* gf   = (const float*)d_in[16];
  const float* bfv  = (const float*)d_in[17];
  // d_in[18] = src_mask (causal triu) applied analytically in attn_kernel.

  char* ws = (char*)d_ws;
  const size_t MB = (size_t)1 << 20;
  float*  pe   = (float*)(ws + 0);          //  4 MB  [2048][512] f32
  bf16_t* Qac  = (bf16_t*)(ws + 4 * MB);    //  4 MB  [b][n][i][d]
  bf16_t* Qbd  = (bf16_t*)(ws + 8 * MB);    //  4 MB
  bf16_t* Kb   = (bf16_t*)(ws + 12 * MB);   //  4 MB
  bf16_t* Vt   = (bf16_t*)(ws + 16 * MB);   //  4 MB  [b][n][d][i]
  bf16_t* Kpos = (bf16_t*)(ws + 20 * MB);   //  2 MB  [n][p][d]
  bf16_t* av   = (bf16_t*)(ws + 22 * MB);   //  4 MB  [s][nd]
  float*  aout = (float*)(ws + 26 * MB);    //  8 MB
  float*  x1   = (float*)(ws + 34 * MB);    //  8 MB
  bf16_t* xb   = (bf16_t*)(ws + 42 * MB);   //  4 MB
  bf16_t* Hb   = (bf16_t*)(ws + 46 * MB);   // 16 MB  [s][FF]
  float*  ffny = (float*)(ws + 62 * MB);    //  8 MB

  posemb_kernel<<<PLEN, 256, 0, stream>>>(pe);

  const dim3 gP(DMODEL / GBN, SEQ / GBM);   // (4, 32)
  gemm_kernel<M_QDUAL, false, false><<<gP, 256, 0, stream>>>(
      src, q_w, rwb, rrb, Qac, Qbd, DMODEL, DMODEL);
  gemm_kernel<M_KHEAD, false, false><<<gP, 256, 0, stream>>>(
      src, k_w, nullptr, nullptr, Kb, nullptr, DMODEL, DMODEL);
  gemm_kernel<M_VHEADT, false, false><<<gP, 256, 0, stream>>>(
      src, v_w, nullptr, nullptr, Vt, nullptr, DMODEL, DMODEL);
  gemm_kernel<M_KPOS, false, false><<<dim3(DMODEL / GBN, PLEN / GBM), 256, 0, stream>>>(
      pe, r_w, nullptr, nullptr, Kpos, nullptr, DMODEL, DMODEL);

  attn_kernel<<<dim3(QLEN / 64, NH, BSZ), 128, 0, stream>>>(Qac, Qbd, Kb, Vt, Kpos, av);

  gemm_kernel<M_F32, true, true><<<gP, 256, 0, stream>>>(
      av, o_w, nullptr, nullptr, aout, nullptr, DMODEL, DMODEL);
  ln_res_kernel<<<SEQ, 128, 0, stream>>>(src, aout, g1, be1, x1, xb);

  gemm_kernel<M_GELU_BF16, true, true><<<dim3(FFD / GBN, SEQ / GBM), 256, 0, stream>>>(
      xb, w1, b1, nullptr, Hb, nullptr, FFD, DMODEL);
  gemm_kernel<M_F32_BIAS, true, true><<<gP, 256, 0, stream>>>(
      Hb, w2, b2, nullptr, ffny, nullptr, DMODEL, FFD);

  ln2_final_kernel<<<SEQ, 128, 0, stream>>>(x1, ffny, g2, be2, gf, bfv, (float*)d_out);
}
